// Gru2_9440338117185
// MI455X (gfx1250) — matrix operand order repaired and checked by reading, not compiled, not run on hardware
//
#include <hip/hip_runtime.h>

// ---- problem constants (match reference) ----
constexpr int CL  = 2;     // layers
constexpr int CHS = 512;   // hidden size
constexpr int CIN = 512;   // input size
constexpr int CB  = 64;    // batch
constexpr int CS  = 1024;  // sequence length

typedef __attribute__((ext_vector_type(16))) __bf16 v16bf;
typedef __attribute__((ext_vector_type(8)))  __bf16 v8bf;
typedef __attribute__((ext_vector_type(4)))  __bf16 v4bf;
typedef __attribute__((ext_vector_type(8)))  float  v8f;
typedef unsigned int u32x4 __attribute__((ext_vector_type(4)));
typedef int          i32x8 __attribute__((ext_vector_type(8)));
typedef int          i32x4 __attribute__((ext_vector_type(4)));

#if defined(__has_builtin)
#  if __has_builtin(__builtin_amdgcn_tensor_load_to_lds)
#    define HAVE_TDM 1
#  endif
#endif

// ---- workspace layout (bytes) ----
constexpr size_t OFF_W     = 0;
constexpr size_t OFF_WH    = 0;                                        // 2 MB
constexpr size_t OFF_WX    = 2u * 1024 * 1024;                         // 2 MB
constexpr size_t OFF_BIAS  = 4u * 1024 * 1024;
constexpr size_t BIAS_EL   = (size_t)CL * 2 * CHS;
constexpr size_t OFF_HB0   = OFF_BIAS + BIAS_EL * 4;
constexpr size_t HB_BYTES  = (size_t)CL * CB * CHS * 2;
constexpr size_t OFF_HB1   = OFF_HB0 + HB_BYTES;
constexpr size_t OFF_HF0   = OFF_HB1 + HB_BYTES;
constexpr size_t HF_BYTES  = (size_t)CL * CB * CHS * 4;
constexpr size_t OFF_HF1   = OFF_HF0 + HF_BYTES;
constexpr size_t OFF_XBF   = OFF_HF1 + HF_BYTES;
constexpr size_t XBF_BYTES = (size_t)CB * CS * CIN * 2;                // 64 MB
constexpr size_t OFF_GX    = OFF_XBF + XBF_BYTES;
constexpr size_t GX_BYTES  = (size_t)CL * 2 * CB * CS * CHS * 4;       // 512 MB
constexpr size_t TOTAL_BIG = OFF_GX + GX_BYTES;

// x-panel staging in LDS (xgemm): 64 t-rows x 1KB, TDM-padded 32B per row
constexpr int PANEL_T    = 64;
constexpr int ROW_BYTES  = CIN * 2;                 // 1024
constexpr int ROW_STRIDE = ROW_BYTES + 32;          // 1056 (264 dw -> 8-bank skew)
constexpr int SMEM_BYTES = PANEL_T * ROW_STRIDE;    // 67584

// A-fragment element mapping (contiguous k order: K = 16*half + e, matching B fills)
__device__ __forceinline__ void afrag_rc(int lane, int e, int& M, int& K) {
    K = 16 * (lane >> 4) + e;
    M = lane & 15;
}

// ============================================================================
// Prep kernels
// ============================================================================
__global__ void prep_wfused(const float* __restrict__ W1, const float* __restrict__ W2,
                            const float* __restrict__ U1, const float* __restrict__ U2,
                            const float* __restrict__ dW1, const float* __restrict__ dW2,
                            const float* __restrict__ dU1, const float* __restrict__ dU2,
                            __bf16* __restrict__ wc) {
    unsigned idx = blockIdx.x * 256u + threadIdx.x;       // < 2*64*32*32*16
    int e = idx & 15, lane = (idx >> 4) & 31, kk = (idx >> 9) & 31;
    int mt = (idx >> 14) & 63, l = idx >> 20;
    int M, K; afrag_rc(lane, e, M, K);
    int row = mt * 16 + M, col = kk * 32 + K;
    int g = (row < CHS) ? 1 : 2, rr = row & (CHS - 1);
    float val;
    if (col < CIN) {
        size_t o = ((size_t)l * CHS + rr) * CIN + col;
        val = W1[l * 3 + g] * W2[o] + dW1[l * 3 + g] * dW2[o];
    } else {
        size_t o = ((size_t)l * CHS + rr) * CHS + (col - CIN);
        val = U1[l * 3 + g] * U2[o] + dU1[l * 3 + g] * dU2[o];
    }
    wc[idx] = (__bf16)val;
}

__global__ void prep_wh(const float* __restrict__ U1, const float* __restrict__ U2,
                        const float* __restrict__ dU1, const float* __restrict__ dU2,
                        __bf16* __restrict__ wh) {
    unsigned idx = blockIdx.x * 256u + threadIdx.x;       // < 2*64*16*32*16
    int e = idx & 15, lane = (idx >> 4) & 31, kk = (idx >> 9) & 15;
    int mt = (idx >> 13) & 63, l = (idx >> 19) & 1;
    int M, K; afrag_rc(lane, e, M, K);
    int row = mt * 16 + M, col = kk * 32 + K;
    int g = (row < CHS) ? 1 : 2, rr = row & (CHS - 1);
    size_t o = ((size_t)l * CHS + rr) * CHS + col;
    wh[idx] = (__bf16)(U1[l * 3 + g] * U2[o] + dU1[l * 3 + g] * dU2[o]);
}

__global__ void prep_wx(const float* __restrict__ W1, const float* __restrict__ W2,
                        const float* __restrict__ dW1, const float* __restrict__ dW2,
                        __bf16* __restrict__ wx) {
    unsigned idx = blockIdx.x * 256u + threadIdx.x;       // < 2*2*32*16*32*16
    int e = idx & 15, lane = (idx >> 4) & 31, kk = (idx >> 9) & 15;
    int mt = (idx >> 13) & 31, zn = (idx >> 18) & 1, l = (idx >> 19) & 1;
    int M, K; afrag_rc(lane, e, M, K);
    int row = mt * 16 + M, col = kk * 32 + K, g = zn + 1;
    size_t o = ((size_t)l * CHS + row) * CIN + col;
    wx[idx] = (__bf16)(W1[l * 3 + g] * W2[o] + dW1[l * 3 + g] * dW2[o]);
}

__global__ void prep_xbf(const float* __restrict__ x, __bf16* __restrict__ xbf) {
    size_t i = ((size_t)blockIdx.x * 256 + threadIdx.x) * 4;
    float4 v = *(const float4*)(x + i);
    v4bf o; o[0] = (__bf16)v.x; o[1] = (__bf16)v.y; o[2] = (__bf16)v.z; o[3] = (__bf16)v.w;
    *(v4bf*)(xbf + i) = o;
}

__global__ void prep_bias(const float* __restrict__ b1, const float* __restrict__ b2,
                          float* __restrict__ bias) {
    int i = blockIdx.x * 256 + threadIdx.x;
    int l = i >> 10, row = i & 1023;
    int g = (row < CHS) ? 1 : 2, rr = row & (CHS - 1);
    size_t o = (size_t)l * 3 * CHS + (size_t)g * CHS + rr;
    bias[i] = b1[o] + b2[o];
}

__global__ void init_h(const float* __restrict__ h0, __bf16* __restrict__ hb,
                       float* __restrict__ hf) {
    int i = blockIdx.x * 256 + threadIdx.x;
    float v = h0[i];
    hb[i] = (__bf16)v;
    hf[i] = v;
}

// ============================================================================
// Big parallel GEMM with TDM-staged x-panel in LDS.
//   Block = one (b, t-chunk) panel (64 KB, one TENSOR_LOAD_TO_LDS) shared by
//   8 waves covering 8 of the 128 (l,zn,mt) row-tiles; 16 blocks per panel.
// ============================================================================
__global__ __launch_bounds__(256) void xgemm(const __bf16* __restrict__ wx,
                                             const __bf16* __restrict__ xbf,
                                             float* __restrict__ gx) {
    extern __shared__ char smem[];
    unsigned bid = blockIdx.x;                 // 16384 blocks
    int panel = bid >> 4, sub = bid & 15;
    int b  = panel >> 4;                       // 0..63
    int tc = panel & 15;                       // 0..15
    int t0 = tc * PANEL_T;
    int w    = threadIdx.x >> 5;
    int lane = threadIdx.x & 31;
    int tid8 = sub * 8 + w;                    // 0..127
    int mt = tid8 & 31, zn = (tid8 >> 5) & 1, l = tid8 >> 6;
    int half = lane >> 4, nl = lane & 15;

    const __bf16* panel_src = xbf + ((size_t)b * CS + t0) * CIN;

#if HAVE_TDM
    if (w == 0) {
        size_t   ga  = (size_t)panel_src;
        unsigned lds = (unsigned)(uintptr_t)smem;      // addr[31:0] == LDS offset
        unsigned n8  = (unsigned)(PANEL_T * ROW_BYTES / 8);   // 8192 x 8B units
        u32x4 g0;
        g0[0] = 1u;                                    // count=1, user mode
        g0[1] = lds;                                   // lds_addr
        g0[2] = (unsigned)(ga & 0xffffffffu);          // global_addr[31:0]
        g0[3] = (unsigned)((ga >> 32) & 0x1ffffffu) | (2u << 30);  // [56:32] | type=2
        i32x8 g1;
        g1[0] = (3 << 16) | (1 << 20) | (7 << 22) | (7 << 25);
        //        data=8B   pad_en      every 256dw   pad 8dw (32B)
        g1[1] = (int)((n8 & 0xffffu) << 16);           // tensor_dim0[15:0]
        g1[2] = (int)(((n8 >> 16) & 0xffffu) | (1u << 16));  // dim0[31:16] | dim1=1
        g1[3] = (int)((n8 & 0xffffu) << 16);           // dim1[31:16]=0 | tile_dim0
        g1[4] = 1;                                     // tile_dim1=1, tile_dim2=0
        g1[5] = (int)n8;                               // dim0_stride[31:0]
        g1[6] = 0;                                     // stride hi, dim1_stride lo
        g1[7] = 0;
        i32x4 z4 = {};
#  if __clang_major__ >= 23
        i32x8 z8 = {};
        __builtin_amdgcn_tensor_load_to_lds(g0, g1, z4, z4, z8, 0);
#  else
        __builtin_amdgcn_tensor_load_to_lds(g0, g1, z4, z4, 0);
#  endif
        __builtin_amdgcn_s_wait_tensorcnt(0);
    }
#else
    // cooperative staged copy with the same 32B-per-row padding
    for (int c = threadIdx.x; c < PANEL_T * (ROW_BYTES / 16); c += 256) {
        int row = c >> 6, col = c & 63;
        *(float4*)(smem + row * ROW_STRIDE + col * 16) =
            *(const float4*)((const char*)panel_src + (size_t)row * ROW_BYTES + col * 16);
    }
#endif
    __syncthreads();

    const __bf16* wa = wx + ((size_t)(((l * 2 + zn) * 32 + mt) * 16)) * 512;

    v8f acc[4] = {};
    for (int kk = 0; kk < 16; ++kk) {
        const __bf16* ap = wa + ((size_t)kk * 32 + lane) * 16;
        if (kk + 1 < 16) __builtin_prefetch(ap + 512, 0, 0);
        v16bf a = *(const v16bf*)ap;
        #pragma unroll
        for (int j = 0; j < 4; ++j) {
            const char* bp = smem + (16 * j + nl) * ROW_STRIDE + kk * 64 + half * 32;
            v16bf bfrag = *(const v16bf*)bp;           // ds_load, 32B aligned
            acc[j] = __builtin_amdgcn_wmma_f32_16x16x32_bf16(false, a, false, bfrag,
                                                             (short)0, acc[j], false, false);
        }
    }
    #pragma unroll
    for (int j = 0; j < 4; ++j) {
        int t = t0 + 16 * j + nl;
        float* gp = gx + (((size_t)(l * 2 + zn) * CB + b) * CS + t) * CHS
                  + mt * 16 + 8 * half;
        *(v8f*)gp = acc[j];
    }
}

// ============================================================================
// Recurrence step, precompute path: K=512 (U@h only) + Gx read in epilogue.
// ============================================================================
__global__ __launch_bounds__(256) void gru_step_pre(
    const __bf16* __restrict__ wh, const float* __restrict__ gx,
    const float* __restrict__ bias,
    const __bf16* __restrict__ hb_prev, const float* __restrict__ hf_prev,
    __bf16* __restrict__ hb_next, float* __restrict__ hf_next,
    float* __restrict__ out_seq, int t) {
    int wid  = blockIdx.x * 8 + (threadIdx.x >> 5);
    int lane = threadIdx.x & 31;
    int l  = wid >> 7, rem = wid & 127;
    int mt = rem >> 2, nt = rem & 3;
    int half = lane >> 4, nl = lane & 15;
    int b = nt * 16 + nl;

    const __bf16* wz = wh + ((size_t)(l * 64 + mt)      * 16) * 512;
    const __bf16* wn = wh + ((size_t)(l * 64 + mt + 32) * 16) * 512;
    const __bf16* hp_base = hb_prev + ((size_t)(l * CB + b)) * CHS;

    v8f accz = {}, accn = {};
    for (int kk = 0; kk < 16; ++kk) {
        v16bf bfrag = *(const v16bf*)(hp_base + kk * 32 + 16 * half);
        v16bf az = *(const v16bf*)(wz + ((size_t)kk * 32 + lane) * 16);
        v16bf an = *(const v16bf*)(wn + ((size_t)kk * 32 + lane) * 16);
        accz = __builtin_amdgcn_wmma_f32_16x16x32_bf16(false, az, false, bfrag,
                                                       (short)0, accz, false, false);
        accn = __builtin_amdgcn_wmma_f32_16x16x32_bf16(false, an, false, bfrag,
                                                       (short)0, accn, false, false);
    }

    int m0 = mt * 16, r0 = m0 + 8 * half;
    size_t hrow = ((size_t)(l * CB + b)) * CHS;
    v8f gz  = *(const v8f*)(gx + (((size_t)(l * 2 + 0) * CB + b) * CS + t) * CHS + r0);
    v8f gn_ = *(const v8f*)(gx + (((size_t)(l * 2 + 1) * CB + b) * CS + t) * CHS + r0);
    v8f bz  = *(const v8f*)(bias + l * 1024 + r0);
    v8f bn  = *(const v8f*)(bias + l * 1024 + CHS + r0);
    v8f hp  = *(const v8f*)(hf_prev + hrow + r0);
    v8f hn; v8bf hnb;
    #pragma unroll
    for (int v = 0; v < 8; ++v) {
        float zg = 1.0f / (1.0f + __expf(-(accz[v] + gz[v] + bz[v])));
        float g  = accn[v] + gn_[v] + bn[v];
        float e2 = __expf(2.0f * g);
        float nn = (e2 - 1.0f) / (e2 + 1.0f);
        float h  = (1.0f - zg) * nn + zg * hp[v];
        hn[v] = h; hnb[v] = (__bf16)h;
    }
    *(v8f*)(hf_next + hrow + r0) = hn;
    *(v8bf*)(hb_next + hrow + r0) = hnb;
    if (l == CL - 1)
        *(v8f*)(out_seq + ((size_t)b * CS + t) * CHS + r0) = hn;
}

// ============================================================================
// Fallback step (small ws): fused K=1024, converts x in-loop.
// ============================================================================
__global__ __launch_bounds__(256) void gru_step_fused(
    const float* __restrict__ x, const __bf16* __restrict__ wc,
    const float* __restrict__ bias,
    const __bf16* __restrict__ hb_prev, const float* __restrict__ hf_prev,
    __bf16* __restrict__ hb_next, float* __restrict__ hf_next,
    float* __restrict__ out_seq, int t) {
    int wid  = blockIdx.x * 8 + (threadIdx.x >> 5);
    int lane = threadIdx.x & 31;
    int l  = wid >> 7, rem = wid & 127;
    int mt = rem >> 2, nt = rem & 3;
    int half = lane >> 4, nl = lane & 15;
    int b = nt * 16 + nl;

    const __bf16* wz = wc + ((size_t)(l * 64 + mt)      * 32) * 512;
    const __bf16* wn = wc + ((size_t)(l * 64 + mt + 32) * 32) * 512;
    const float*  xb = x + ((size_t)b * CS + t) * CIN;
    const __bf16* hp_base = hb_prev + ((size_t)(l * CB + b)) * CHS;

    v8f accz = {}, accn = {};
    for (int kk = 0; kk < 32; ++kk) {
        v16bf bfrag;
        if (kk < 16) {
            const float* xp = xb + kk * 32 + 16 * half;
            #pragma unroll
            for (int e = 0; e < 16; ++e) bfrag[e] = (__bf16)xp[e];
        } else {
            bfrag = *(const v16bf*)(hp_base + (kk - 16) * 32 + 16 * half);
        }
        v16bf az = *(const v16bf*)(wz + ((size_t)kk * 32 + lane) * 16);
        v16bf an = *(const v16bf*)(wn + ((size_t)kk * 32 + lane) * 16);
        accz = __builtin_amdgcn_wmma_f32_16x16x32_bf16(false, az, false, bfrag,
                                                       (short)0, accz, false, false);
        accn = __builtin_amdgcn_wmma_f32_16x16x32_bf16(false, an, false, bfrag,
                                                       (short)0, accn, false, false);
    }

    int m0 = mt * 16, r0 = m0 + 8 * half;
    size_t hrow = ((size_t)(l * CB + b)) * CHS;
    v8f bz = *(const v8f*)(bias + l * 1024 + r0);
    v8f bn = *(const v8f*)(bias + l * 1024 + CHS + r0);
    v8f hp = *(const v8f*)(hf_prev + hrow + r0);
    v8f hn; v8bf hnb;
    #pragma unroll
    for (int v = 0; v < 8; ++v) {
        float zg = 1.0f / (1.0f + __expf(-(accz[v] + bz[v])));
        float g  = accn[v] + bn[v];
        float e2 = __expf(2.0f * g);
        float nn = (e2 - 1.0f) / (e2 + 1.0f);
        float h  = (1.0f - zg) * nn + zg * hp[v];
        hn[v] = h; hnb[v] = (__bf16)h;
    }
    *(v8f*)(hf_next + hrow + r0) = hn;
    *(v8bf*)(hb_next + hrow + r0) = hnb;
    if (l == CL - 1)
        *(v8f*)(out_seq + ((size_t)b * CS + t) * CHS + r0) = hn;
}

__global__ void write_state(const float* __restrict__ hf, float* __restrict__ out_state) {
    int i = blockIdx.x * 256 + threadIdx.x;
    out_state[i] = hf[i];
}

// ============================================================================
extern "C" void kernel_launch(void* const* d_in, const int* in_sizes, int n_in,
                              void* d_out, int out_size, void* d_ws, size_t ws_size,
                              hipStream_t stream) {
    const float* x   = (const float*)d_in[0];
    const float* h0  = (const float*)d_in[1];
    const float* W1  = (const float*)d_in[2];
    const float* W2  = (const float*)d_in[3];
    const float* U1  = (const float*)d_in[4];
    const float* U2  = (const float*)d_in[5];
    const float* dW1 = (const float*)d_in[6];
    const float* dW2 = (const float*)d_in[7];
    const float* dU1 = (const float*)d_in[8];
    const float* dU2 = (const float*)d_in[9];
    const float* b1  = (const float*)d_in[10];
    const float* b2  = (const float*)d_in[11];

    char* ws = (char*)d_ws;
    float*  bias  = (float*)(ws + OFF_BIAS);
    __bf16* hb[2] = { (__bf16*)(ws + OFF_HB0), (__bf16*)(ws + OFF_HB1) };
    float*  hf[2] = { (float*)(ws + OFF_HF0),  (float*)(ws + OFF_HF1)  };

    float* out_seq   = (float*)d_out;
    float* out_state = (float*)d_out + (size_t)CB * CS * CHS;

    prep_bias<<<(unsigned)(BIAS_EL / 256), 256, 0, stream>>>(b1, b2, bias);
    init_h<<<(CL * CB * CHS) / 256, 256, 0, stream>>>(h0, hb[0], hf[0]);

    if (ws_size >= TOTAL_BIG) {
        __bf16* wh  = (__bf16*)(ws + OFF_WH);
        __bf16* wxp = (__bf16*)(ws + OFF_WX);
        __bf16* xbf = (__bf16*)(ws + OFF_XBF);
        float*  gx  = (float*)(ws + OFF_GX);
        prep_wh<<<4096, 256, 0, stream>>>(U1, U2, dU1, dU2, wh);
        prep_wx<<<4096, 256, 0, stream>>>(W1, W2, dW1, dW2, wxp);
        prep_xbf<<<(unsigned)(((size_t)CB * CS * CIN / 4) / 256), 256, 0, stream>>>(x, xbf);
        xgemm<<<16384, 256, SMEM_BYTES, stream>>>(wxp, xbf, gx);
        for (int t = 0; t < CS; ++t) {
            int cur = t & 1, nxt = cur ^ 1;
            gru_step_pre<<<32, 256, 0, stream>>>(wh, gx, bias, hb[cur], hf[cur],
                                                 hb[nxt], hf[nxt], out_seq, t);
        }
    } else {
        __bf16* wc = (__bf16*)(ws + OFF_W);
        prep_wfused<<<8192, 256, 0, stream>>>(W1, W2, U1, U2, dW1, dW2, dU1, dU2, wc);
        for (int t = 0; t < CS; ++t) {
            int cur = t & 1, nxt = cur ^ 1;
            gru_step_fused<<<32, 256, 0, stream>>>(x, wc, bias, hb[cur], hf[cur],
                                                   hb[nxt], hf[nxt], out_seq, t);
        }
    }
    write_state<<<(CL * CB * CHS) / 256, 256, 0, stream>>>(hf[CS & 1], out_state);
}
